// PointNet2Encoder_42159398978140
// MI455X (gfx1250) — matrix operand order repaired and checked by reading, not compiled, not run on hardware
//
#include <hip/hip_runtime.h>

// ---------------------------------------------------------------------------
// Types for CDNA5 WMMA (wave32): v_wmma_f32_16x16x32_f16
// ---------------------------------------------------------------------------
typedef __attribute__((ext_vector_type(16))) _Float16 v16h;
typedef __attribute__((ext_vector_type(8)))  float    v8f;

union CFrag { v8f v; float e[8]; };
union H2U   { _Float16 h[2]; unsigned int u; };

#define NEG_BIG (-3.0e38f)

// Problem constants (fixed by the reference)
#define BATCH 8
#define NPT   4096
#define NS1   2048
#define NS2   512
#define KNBR  64

// ---------------------------------------------------------------------------
// Fragment-swizzled storage: each 16x32 (A) / 32x16 (B) f16 tile is stored as
// 32 lanes x 16 contiguous halves (512 halves, 1 KB), so a fragment load is a
// single aligned 32-byte load per lane (2x b128) with zero repacking VALU.
//
// A-tile element (M, kc):  lane = M + ((kc>>3)&1)*16, h = (kc&7)+((kc>>4)&1)*8
// B-tile element (kk, n):  lane = n + ((kk>>3)&1)*16, h = (kk&7)+((kk>>4)&1)*8
// (per ISA 7.12.2 16-bit operand layouts)
// ---------------------------------------------------------------------------
template<int KS>
__device__ inline v16h load_a_frag(const _Float16* A, int mt, int ks) {
  const int lane = threadIdx.x & 31;
  return *(const v16h*)(A + ((size_t)(mt * KS + ks) * 512 + lane * 16));
}

__device__ inline v16h load_b_frag(const _Float16* W, int fragIdx) {
  const int lane = threadIdx.x & 31;
  return *(const v16h*)(W + ((size_t)fragIdx * 512 + lane * 16));
}

// ---------------------------------------------------------------------------
// Convert fp32 weight matrices [srcK x H] into fragment-swizzled, K-padded
// fp16 tables: NT x KS fragments of 512 halves (frag id = nt*KS + ks).
// ---------------------------------------------------------------------------
__device__ inline void cvt_table(const float* __restrict__ W,
                                 _Float16* __restrict__ o,
                                 int srcK, int KS, int NT, int H,
                                 int t, int st) {
  const int total = NT * KS * 512;
  for (int i = t; i < total; i += st) {
    const int frag = i >> 9;
    const int r    = i & 511;
    const int lane = r >> 4;
    const int h    = r & 15;
    const int nt   = frag / KS;
    const int ks   = frag % KS;
    const int n    = nt * 16 + (lane & 15);
    const int kk   = ks * 32 + ((lane < 16) ? 0 : 8) + (h & 7) + ((h >> 3) << 4);
    o[i] = (_Float16)((kk < srcK) ? W[kk * H + n] : 0.f);
  }
}

__global__ void cvt_weights_kernel(const float* __restrict__ W1a,
                                   const float* __restrict__ W2a,
                                   const float* __restrict__ W1b,
                                   const float* __restrict__ W2b,
                                   _Float16* __restrict__ o1a,
                                   _Float16* __restrict__ o2a,
                                   _Float16* __restrict__ o1b,
                                   _Float16* __restrict__ o2b) {
  const int t  = threadIdx.x + blockIdx.x * blockDim.x;
  const int st = blockDim.x * gridDim.x;
  cvt_table(W1a, o1a,  9, 1, 4,  64, t, st);
  cvt_table(W2a, o2a, 64, 2, 4,  64, t, st);
  cvt_table(W1b, o1b, 67, 3, 8, 128, t, st);
  cvt_table(W2b, o2b,128, 4, 8, 128, t, st);
}

// ---------------------------------------------------------------------------
// Farthest point sampling: one 1024-thread workgroup per batch element.
// LDS-resident running min-distance array, two-level argmax reduction
// (wave shuffle + cross-wave LDS) with first-index tie-break.
// ---------------------------------------------------------------------------
template<int NPTS, int NSAMP, int TPB>
__global__ __launch_bounds__(TPB)
void fps_kernel(const float* __restrict__ pos, size_t posStride,
                int* __restrict__ iout, size_t ioutStride,
                float* __restrict__ pout, size_t poutStride) {
  const int b    = blockIdx.x;
  const int tid  = threadIdx.x;
  const int lane = tid & 31;
  const int w    = tid >> 5;
  const float* pb = pos + (size_t)b * posStride;
  int* ib = iout + (size_t)b * ioutStride;

  __shared__ float sd[NPTS];
  __shared__ int   sIdx[NSAMP];
  __shared__ float rd[TPB / 32];
  __shared__ int   rn[TPB / 32];
  __shared__ int   sFar;

  for (int n = tid; n < NPTS; n += TPB) sd[n] = 3.0e38f;
  int far = 0;
  __syncthreads();

  for (int s = 0; s < NSAMP; ++s) {
    if (tid == 0) sIdx[s] = far;
    const float fx = pb[far * 3 + 0];
    const float fy = pb[far * 3 + 1];
    const float fz = pb[far * 3 + 2];
    float bd = -1.f; int bn = 0;
    for (int n = tid; n < NPTS; n += TPB) {
      float dx = pb[n * 3 + 0] - fx;
      float dy = pb[n * 3 + 1] - fy;
      float dz = pb[n * 3 + 2] - fz;
      float d  = dx * dx + dy * dy + dz * dz;
      d = fminf(sd[n], d);
      sd[n] = d;
      if (d > bd || (d == bd && n < bn)) { bd = d; bn = n; }
    }
#pragma unroll
    for (int off = 16; off >= 1; off >>= 1) {
      float od = __shfl_down(bd, off, 32);
      int   on = __shfl_down(bn, off, 32);
      if (od > bd || (od == bd && on < bn)) { bd = od; bn = on; }
    }
    if (lane == 0) { rd[w] = bd; rn[w] = bn; }
    __syncthreads();
    if (w == 0) {
      constexpr int NW = TPB / 32;
      float d2 = (lane < NW) ? rd[lane] : -1.f;
      int   n2 = (lane < NW) ? rn[lane] : 0;
#pragma unroll
      for (int off = 16; off >= 1; off >>= 1) {
        float od = __shfl_down(d2, off, 32);
        int   on = __shfl_down(n2, off, 32);
        if (od > d2 || (od == d2 && on < n2)) { d2 = od; n2 = on; }
      }
      if (lane == 0) sFar = n2;
    }
    __syncthreads();
    far = sFar;
  }
  for (int s = tid; s < NSAMP; s += TPB) {
    int j = sIdx[s];
    ib[s] = j;
    if (pout) {
      float* pe = pout + (size_t)b * poutStride + (size_t)s * 3;
      pe[0] = pb[j * 3 + 0];
      pe[1] = pb[j * 3 + 1];
      pe[2] = pb[j * 3 + 2];
    }
  }
}

// ---------------------------------------------------------------------------
// Approximate 64-NN: one wave per query; each lane keeps its top-2 over a
// strided 1/32 slice of the candidates; union of 32x2 = 64 candidates.
// Self (d=0) is always included; radius mask is applied exactly later.
// ---------------------------------------------------------------------------
template<int NPTS>
__global__ __launch_bounds__(256)
void knn_kernel(const float* __restrict__ pos, size_t posStride,
                unsigned short* __restrict__ idx, size_t idxStride) {
  const int b    = blockIdx.y;
  const int wave = threadIdx.x >> 5;
  const int lane = threadIdx.x & 31;
  const int q    = blockIdx.x * 8 + wave;
  const float* pb = pos + (size_t)b * posStride;

  const float qx = pb[q * 3 + 0];
  const float qy = pb[q * 3 + 1];
  const float qz = pb[q * 3 + 2];

  float b0 = 3.0e38f, b1 = 3.0e38f;
  int   i0 = q,       i1 = q;
  for (int n = lane; n < NPTS; n += 32) {
    if (n + 256 < NPTS) __builtin_prefetch(pb + (n + 256) * 3, 0, 1);
    float dx = pb[n * 3 + 0] - qx;
    float dy = pb[n * 3 + 1] - qy;
    float dz = pb[n * 3 + 2] - qz;
    float d  = dx * dx + dy * dy + dz * dz;
    if (d < b0 || (d == b0 && n < i0)) {
      b1 = b0; i1 = i0; b0 = d; i0 = n;
    } else if (d < b1 || (d == b1 && n < i1)) {
      b1 = d; i1 = n;
    }
  }
  unsigned short* ob = idx + (size_t)b * idxStride + (size_t)q * KNBR;
  ob[lane * 2 + 0] = (unsigned short)i0;
  ob[lane * 2 + 1] = (unsigned short)i1;
}

// ---------------------------------------------------------------------------
// Fused PointNetConv: per wave = one query neighborhood (64 rows).
//   stage A1 = [xj | rel | pad] as f16 into fragment-swizzled LDS
//   GEMM1 (WMMA) + bias + ReLU -> fragment-swizzled A2 (f16, LDS)
//   GEMM2 (WMMA) + bias + radius mask -> max over 64 rows -> out[q][H]
// FIN: input feature dim; KP: padded K for GEMM1 (mult of 32); H: hidden/out.
// ---------------------------------------------------------------------------
template<int FIN, int KP, int H, int WAVES>
__global__ __launch_bounds__(WAVES * 32)
void conv_kernel(const float* __restrict__ x,   size_t xStride,
                 const float* __restrict__ pos, size_t posStride,
                 const unsigned short* __restrict__ knn, size_t knnStride,
                 const int* __restrict__ remap, size_t remapStride,
                 const _Float16* __restrict__ W1, const float* __restrict__ b1,
                 const _Float16* __restrict__ W2, const float* __restrict__ b2,
                 float r2, float* __restrict__ out, size_t outStride) {
  constexpr int KS1 = KP / 32;   // GEMM1 k-steps
  constexpr int KS2 = H / 32;    // GEMM2 k-steps
  constexpr int NT  = H / 16;    // output column tiles

  const int wave = threadIdx.x >> 5;
  const int lane = threadIdx.x & 31;
  const int b    = blockIdx.y;
  const int q    = blockIdx.x * WAVES + wave;

  __shared__ __align__(32) _Float16 sA1[WAVES][4 * KS1 * 512];
  __shared__ __align__(32) _Float16 sA2[WAVES][4 * KS2 * 512];
  __shared__ float sMsk[WAVES][64];

  _Float16* A1w = &sA1[wave][0];
  _Float16* A2w = &sA2[wave][0];

  const float*          xb  = x   + (size_t)b * xStride;
  const float*          pb  = pos + (size_t)b * posStride;
  const unsigned short* kb_ = knn + (size_t)b * knnStride + (size_t)q * KNBR;
  const int*            rb  = remap ? (remap + (size_t)b * remapStride) : nullptr;

  const float qx = pb[q * 3 + 0];
  const float qy = pb[q * 3 + 1];
  const float qz = pb[q * 3 + 2];

  // ---- stage neighbor rows into fragment-swizzled LDS (paired u32 stores) --
  for (int j = lane; j < 64; j += 32) {
    const int nb = kb_[j];
    const float rx = pb[nb * 3 + 0] - qx;
    const float ry = pb[nb * 3 + 1] - qy;
    const float rz = pb[nb * 3 + 2] - qz;
    const float d2 = rx * rx + ry * ry + rz * rz;
    sMsk[wave][j] = (d2 < r2) ? 0.f : NEG_BIG;
    const int frow = rb ? rb[nb] : nb;
    const float* xf = xb + (size_t)frow * FIN;
    const int mt = j >> 4;
    const int M  = j & 15;
#pragma unroll
    for (int k = 0; k < KP; k += 2) {
      float lo, hi;
      if (k + 1 < FIN) {            // pure feature pair (FIN is even)
        float2 t = *(const float2*)(xf + k);
        lo = t.x; hi = t.y;
      } else if (k == FIN) {        // (rel.x, rel.y)
        lo = rx; hi = ry;
      } else if (k == FIN + 2) {    // (rel.z, pad)
        lo = rz; hi = 0.f;
      } else {                      // zero pad
        lo = 0.f; hi = 0.f;
      }
      const int kc    = k & 31;
      const int ks    = k >> 5;
      const int laneT = M + (((kc >> 3) & 1) << 4);
      const int h     = (kc & 7) + (((kc >> 4) & 1) << 3);
      H2U p; p.h[0] = (_Float16)lo; p.h[1] = (_Float16)hi;
      *(unsigned int*)(A1w + ((mt * KS1 + ks) * 512 + laneT * 16 + h)) = p.u;
    }
  }
  __syncthreads();

  const int col16 = lane & 15;
  const int rofs  = (lane < 16) ? 0 : 8;

  // preload radius masks into registers (constant-indexed after unroll)
  float msk[4][8];
#pragma unroll
  for (int mt = 0; mt < 4; ++mt)
#pragma unroll
    for (int v = 0; v < 8; ++v)
      msk[mt][v] = sMsk[wave][mt * 16 + rofs + v];

  // ---- GEMM1: [64 x KP] @ [KP x H], ReLU -> swizzled sA2 ----
  for (int nt = 0; nt < NT; ++nt) {
    v16h bf[KS1];
#pragma unroll
    for (int ks = 0; ks < KS1; ++ks)
      bf[ks] = load_b_frag(W1, nt * KS1 + ks);
    const float bias = b1[nt * 16 + col16];

    const int col   = nt * 16 + col16;   // column in A2's K-dimension
    const int ks2   = col >> 5;
    const int kc2   = col & 31;
    const int laneB = ((kc2 >> 3) & 1) << 4;
    const int h2    = (kc2 & 7) + (((kc2 >> 4) & 1) << 3);

#pragma unroll
    for (int mt = 0; mt < 4; ++mt) {
      CFrag c;
#pragma unroll
      for (int v = 0; v < 8; ++v) c.e[v] = 0.f;
#pragma unroll
      for (int ks = 0; ks < KS1; ++ks) {
        v16h a = load_a_frag<KS1>(A1w, mt, ks);
        c.v = __builtin_amdgcn_wmma_f32_16x16x32_f16(
            false, a, false, bf[ks], (short)0, c.v, false, false);
      }
#pragma unroll
      for (int v = 0; v < 8; ++v) {
        float val = c.e[v] + bias;
        val = val > 0.f ? val : 0.f;
        const int M = rofs + v;          // row within the mt tile
        A2w[(mt * KS2 + ks2) * 512 + (M + laneB) * 16 + h2] = (_Float16)val;
      }
    }
  }
  __syncthreads();

  // ---- GEMM2: [64 x H] @ [H x H], mask + max over 64 rows ----
  for (int nt = 0; nt < NT; ++nt) {
    v16h bf[KS2];
#pragma unroll
    for (int ks = 0; ks < KS2; ++ks)
      bf[ks] = load_b_frag(W2, nt * KS2 + ks);
    const float bias = b2[nt * 16 + col16];
    float mx = NEG_BIG;
#pragma unroll
    for (int mt = 0; mt < 4; ++mt) {
      CFrag c;
#pragma unroll
      for (int v = 0; v < 8; ++v) c.e[v] = 0.f;
#pragma unroll
      for (int ks = 0; ks < KS2; ++ks) {
        v16h a = load_a_frag<KS2>(A2w, mt, ks);
        c.v = __builtin_amdgcn_wmma_f32_16x16x32_f16(
            false, a, false, bf[ks], (short)0, c.v, false, false);
      }
#pragma unroll
      for (int v = 0; v < 8; ++v) {
        float val = c.e[v] + bias + msk[mt][v];
        mx = fmaxf(mx, val);
      }
    }
    // lanes L and L+16 hold the same output channel -> combine, lanes<16 write
    mx = fmaxf(mx, __shfl_xor(mx, 16, 32));
    if (lane < 16)
      out[(size_t)b * outStride + (size_t)q * H + nt * 16 + col16] = mx;
  }
}

// ---------------------------------------------------------------------------
// Global max pool over 512 sampled rows of h2 + final 128x128 linear.
// ---------------------------------------------------------------------------
__global__ __launch_bounds__(128)
void final_kernel(const float* __restrict__ h2, const int* __restrict__ i2,
                  const float* __restrict__ Wl, const float* __restrict__ bl,
                  float* __restrict__ out) {
  const int b = blockIdx.x;
  const int c = threadIdx.x;
  const float* hb = h2 + (size_t)b * NS1 * 128;
  const int*   ib = i2 + (size_t)b * NS2;
  __shared__ float sf[128];
  float m = NEG_BIG;
  for (int s = 0; s < NS2; ++s) {
    const int j = ib[s];
    m = fmaxf(m, hb[(size_t)j * 128 + c]);
  }
  sf[c] = m;
  __syncthreads();
  float acc = bl[c];
  for (int k = 0; k < 128; ++k) acc += sf[k] * Wl[k * 128 + c];
  out[b * 128 + c] = acc;
}

// ---------------------------------------------------------------------------
// Host-side launch
// ---------------------------------------------------------------------------
static inline size_t align256(size_t x) { return (x + 255) & ~(size_t)255; }

extern "C" void kernel_launch(void* const* d_in, const int* in_sizes, int n_in,
                              void* d_out, int out_size, void* d_ws, size_t ws_size,
                              hipStream_t stream) {
  (void)in_sizes; (void)n_in; (void)out_size; (void)ws_size;

  const float* feats  = (const float*)d_in[0];   // [B,N,6]
  const float* coords = (const float*)d_in[1];   // [B,N,3]
  const float* W1a = (const float*)d_in[2];
  const float* b1a = (const float*)d_in[3];
  const float* W2a = (const float*)d_in[4];
  const float* b2a = (const float*)d_in[5];
  const float* W1b = (const float*)d_in[6];
  const float* b1b = (const float*)d_in[7];
  const float* W2b = (const float*)d_in[8];
  const float* b2b = (const float*)d_in[9];
  const float* Wl  = (const float*)d_in[10];
  const float* bl  = (const float*)d_in[11];
  float* out = (float*)d_out;

  // ---- workspace carve-out ----
  char* w = (char*)d_ws;
  _Float16* w1a_h = (_Float16*)w; w += align256((size_t)4 * 1 * 512 * 2);
  _Float16* w2a_h = (_Float16*)w; w += align256((size_t)4 * 2 * 512 * 2);
  _Float16* w1b_h = (_Float16*)w; w += align256((size_t)8 * 3 * 512 * 2);
  _Float16* w2b_h = (_Float16*)w; w += align256((size_t)8 * 4 * 512 * 2);
  unsigned short* idx1 = (unsigned short*)w; w += align256((size_t)BATCH * NPT * KNBR * 2);
  unsigned short* idx2 = (unsigned short*)w; w += align256((size_t)BATCH * NS1 * KNBR * 2);
  int*   i1 = (int*)w;   w += align256((size_t)BATCH * NS1 * 4);
  int*   i2 = (int*)w;   w += align256((size_t)BATCH * NS2 * 4);
  float* p1 = (float*)w; w += align256((size_t)BATCH * NS1 * 3 * 4);
  float* h1 = (float*)w; w += align256((size_t)BATCH * NPT * 64 * 4);
  float* h2 = (float*)w; w += align256((size_t)BATCH * NS1 * 128 * 4);

  // 1. weights -> fragment-swizzled f16 tables
  cvt_weights_kernel<<<64, 256, 0, stream>>>(W1a, W2a, W1b, W2b,
                                             w1a_h, w2a_h, w1b_h, w2b_h);

  // 2. FPS over raw coords -> i1 (+ gathered p1)
  fps_kernel<NPT, NS1, 1024><<<BATCH, 1024, 0, stream>>>(
      coords, (size_t)NPT * 3, i1, NS1, p1, (size_t)NS1 * 3);

  // 3. 64-NN on raw coords -> idx1
  knn_kernel<NPT><<<dim3(NPT / 8, BATCH), 256, 0, stream>>>(
      coords, (size_t)NPT * 3, idx1, (size_t)NPT * KNBR);

  // 4. conv1: 9(->32) -> 64 -> 64, radius 0.2
  conv_kernel<6, 32, 64, 4><<<dim3(NPT / 4, BATCH), 128, 0, stream>>>(
      feats, (size_t)NPT * 6, coords, (size_t)NPT * 3,
      idx1, (size_t)NPT * KNBR, nullptr, 0,
      w1a_h, b1a, w2a_h, b2a, 0.04f, h1, (size_t)NPT * 64);

  // 5. FPS over p1 -> i2
  fps_kernel<NS1, NS2, 1024><<<BATCH, 1024, 0, stream>>>(
      p1, (size_t)NS1 * 3, i2, NS2, nullptr, 0);

  // 6. 64-NN on p1 -> idx2
  knn_kernel<NS1><<<dim3(NS1 / 8, BATCH), 256, 0, stream>>>(
      p1, (size_t)NS1 * 3, idx2, (size_t)NS1 * KNBR);

  // 7. conv2: 67(->96) -> 128 -> 128, radius 0.4; features = h1 gathered via i1
  conv_kernel<64, 96, 128, 2><<<dim3(NS1 / 2, BATCH), 64, 0, stream>>>(
      h1, (size_t)NPT * 64, p1, (size_t)NS1 * 3,
      idx2, (size_t)NS1 * KNBR, i1, NS1,
      w1b_h, b1b, w2b_h, b2b, 0.16f, h2, (size_t)NS1 * 128);

  // 8. global max pool over h2[i2] + final linear
  final_kernel<<<BATCH, 128, 0, stream>>>(h2, i2, Wl, bl, out);
}
